// Attention_aggregator_23373212025263
// MI455X (gfx1250) — matrix operand order repaired and checked by reading, not compiled, not run on hardware
//
#include <hip/hip_runtime.h>
#include <hip/hip_bf16.h>

#define B_ 8
#define N_ 4096
#define F_ 128
#define E_ 128
#define NEGV (-9999999.0f)

typedef __attribute__((ext_vector_type(16))) _Float16 v16h;
typedef __attribute__((ext_vector_type(8)))  _Float16 v8h;
typedef __attribute__((ext_vector_type(8)))  float    v8f;

__device__ __forceinline__ v16h join8(v8h lo, v8h hi) {
  return __builtin_shufflevector(lo, hi, 0,1,2,3,4,5,6,7,8,9,10,11,12,13,14,15);
}

__device__ __forceinline__ v8f wmma_f16(v16h a, v16h b, v8f c) {
  // D = A(16x32 f16) * B(32x16 f16) + C(16x16 f32)
  return __builtin_amdgcn_wmma_f32_16x16x32_f16(false, a, false, b, (short)0, c,
                                                false, false);
}

// ---------------------------------------------------------------------------
// Kernel 1: fp32 -> f16, row-major (Xh) and transposed (Xt) copies.
// Xh feeds A-fragments (Q rows) and S-GEMM B-fragments (K rows, f-contiguous).
// Xt feeds PV-GEMM B-fragments (column/key-contiguous per feature).
// ---------------------------------------------------------------------------
__global__ void cvt_x_kernel(const float* __restrict__ X,
                             _Float16* __restrict__ Xh,
                             _Float16* __restrict__ Xt) {
  size_t i = (size_t)blockIdx.x * blockDim.x + threadIdx.x;
  size_t total = (size_t)B_ * N_ * F_;
  if (i >= total) return;
  int f = (int)(i % F_);
  int r = (int)((i / F_) % N_);
  int b = (int)(i / ((size_t)N_ * F_));
  float v = X[i];
  Xh[i] = (_Float16)v;
  Xt[((size_t)b * F_ + f) * N_ + r] = (_Float16)v;
}

__global__ void cvt_w_kernel(const float* __restrict__ W,
                             _Float16* __restrict__ Wh) {
  int i = blockIdx.x * blockDim.x + threadIdx.x;
  if (i < E_ * 2 * F_) Wh[i] = (_Float16)W[i];
}

// ---------------------------------------------------------------------------
// Kernel 2: masked-softmax attention, flash style. One wave32 per 16-row block.
// ---------------------------------------------------------------------------
__global__ __launch_bounds__(256) void attn_kernel(
    const _Float16* __restrict__ Xh, const _Float16* __restrict__ Xt,
    const int* __restrict__ adj, _Float16* __restrict__ Ch) {
  __shared__ _Float16 ldsP[8 * 16 * 32];  // 1KB P-transpose buffer per wave

  const int lane = threadIdx.x & 31;
  const int warp = threadIdx.x >> 5;
  const int w    = blockIdx.x * 8 + warp;          // global wave id
  const int b    = w / (N_ / 16);
  const int row0 = (w % (N_ / 16)) * 16;
  const int l16  = lane & 15;
  const int h    = lane >> 4;
  _Float16* P = ldsP + warp * 512;

  const _Float16* Xb  = Xh  + (size_t)b * N_ * F_;
  const _Float16* Xtb = Xt  + (size_t)b * F_ * N_;
  const int*      Ab  = adj + (size_t)b * N_ * N_;

  // Resident Q A-fragments: lane -> row row0+l16; per-lane K contiguous (k = 16h + s), same order as the B loads
  v16h qa[4];
  {
    const _Float16* qrow = Xb + (size_t)(row0 + l16) * F_;
#pragma unroll
    for (int kc = 0; kc < 4; ++kc) {
      v8h lo = *(const v8h*)(qrow + kc * 32 + 16 * h);
      v8h hi = *(const v8h*)(qrow + kc * 32 + 16 * h + 8);
      qa[kc] = join8(lo, hi);
    }
  }

  v8f zero = {};
  v8f o[8];
#pragma unroll
  for (int t = 0; t < 8; ++t) o[t] = zero;
  float m[8], sa[8];
#pragma unroll
  for (int v = 0; v < 8; ++v) { m[v] = NEGV; sa[v] = 0.0f; }

  for (int c0 = 0; c0 < N_; c0 += 32) {
    // Prefetch next chunk's mask rows
    if (c0 + 32 < N_)
      __builtin_prefetch(Ab + (size_t)(row0 + l16) * N_ + c0 + 32, 0, 1);

    // ---- S tiles: two 16-col tiles, K = F accumulated over 4 WMMAs each ----
    v8f s0 = zero, s1 = zero;
#pragma unroll
    for (int kc = 0; kc < 4; ++kc) {
      v16h b0 = *(const v16h*)(Xb + (size_t)(c0 + l16) * F_ + kc * 32 + 16 * h);
      v16h b1 = *(const v16h*)(Xb + (size_t)(c0 + 16 + l16) * F_ + kc * 32 + 16 * h);
      s0 = wmma_f16(qa[kc], b0, s0);
      s1 = wmma_f16(qa[kc], b1, s1);
    }

    // ---- mask (C layout: VGPR v, half h -> row v+8h; lane%16 -> col) ----
#pragma unroll
    for (int v = 0; v < 8; ++v) {
      const int* ar = Ab + (size_t)(row0 + v + 8 * h) * N_ + c0 + l16;
      int a0 = ar[0];
      int a1 = ar[16];
      s0[v] = a0 ? s0[v] : NEGV;
      s1[v] = a1 ? s1[v] : NEGV;
    }

    // ---- online softmax: row max across 16 lanes, rescale, exp ----
    v8f scv;
#pragma unroll
    for (int v = 0; v < 8; ++v) {
      float tm = fmaxf(s0[v], s1[v]);
      tm = fmaxf(tm, __shfl_xor(tm, 1, 32));
      tm = fmaxf(tm, __shfl_xor(tm, 2, 32));
      tm = fmaxf(tm, __shfl_xor(tm, 4, 32));
      tm = fmaxf(tm, __shfl_xor(tm, 8, 32));
      float mn = fmaxf(m[v], tm);
      float sc = __expf(m[v] - mn);
      m[v] = mn;
      float p0 = __expf(s0[v] - mn);
      float p1 = __expf(s1[v] - mn);
      sa[v] = sa[v] * sc + p0 + p1;   // per-lane partial row sums (reduced once at end)
      s0[v] = p0;
      s1[v] = p1;
      scv[v] = sc;
    }
#pragma unroll
    for (int t = 0; t < 8; ++t) o[t] *= scv;  // rescale accumulator

    // ---- P: C layout -> A layout via per-wave LDS region ----
#pragma unroll
    for (int v = 0; v < 8; ++v) {
      P[(v + 8 * h) * 32 + l16]      = (_Float16)s0[v];
      P[(v + 8 * h) * 32 + 16 + l16] = (_Float16)s1[v];
    }
    asm volatile("s_wait_dscnt 0x0" ::: "memory");  // wave-private LDS: counter wait only
    v8h plo = *(const v8h*)(P + l16 * 32 + 16 * h);
    v8h phi = *(const v8h*)(P + l16 * 32 + 16 * h + 8);
    v16h pa = join8(plo, phi);

    // ---- O += P(16x32) * V(32x128): B-fragments contiguous from Xt ----
#pragma unroll
    for (int ft = 0; ft < 8; ++ft) {
      v16h bv = *(const v16h*)(Xtb + (size_t)(ft * 16 + l16) * N_ + c0 + 16 * h);
      o[ft] = wmma_f16(pa, bv, o[ft]);
    }
  }

  // ---- final row-sum reduction and normalization ----
  v8f rinv;
#pragma unroll
  for (int v = 0; v < 8; ++v) {
    float s = sa[v];
    s += __shfl_xor(s, 1, 32);
    s += __shfl_xor(s, 2, 32);
    s += __shfl_xor(s, 4, 32);
    s += __shfl_xor(s, 8, 32);
    rinv[v] = 1.0f / s;
  }

  _Float16* Cb = Ch + (size_t)b * N_ * F_;
#pragma unroll
  for (int ft = 0; ft < 8; ++ft) {
    v8f ov = o[ft] * rinv;
#pragma unroll
    for (int v = 0; v < 8; ++v) {
      Cb[(size_t)(row0 + v + 8 * h) * F_ + ft * 16 + l16] = (_Float16)ov[v];
    }
  }
}

// ---------------------------------------------------------------------------
// Kernel 3: out = relu([Xh | Ch] @ W^T).  W row-major [E, 2F] is already the
// correct B-matrix memory order (K-contiguous per output column).
// ---------------------------------------------------------------------------
__global__ __launch_bounds__(256) void linear_kernel(
    const _Float16* __restrict__ Xh, const _Float16* __restrict__ Ch,
    const _Float16* __restrict__ Wh, float* __restrict__ out) {
  const int lane = threadIdx.x & 31;
  const int warp = threadIdx.x >> 5;
  const int w    = blockIdx.x * 8 + warp;
  const int b    = w / (N_ / 16);
  const int row0 = (w % (N_ / 16)) * 16;
  const int l16  = lane & 15;
  const int h    = lane >> 4;

  v8f zero = {};
  v8f acc[8];
#pragma unroll
  for (int t = 0; t < 8; ++t) acc[t] = zero;

  const _Float16* xrow = Xh + ((size_t)b * N_ + row0 + l16) * F_;
  const _Float16* crow = Ch + ((size_t)b * N_ + row0 + l16) * F_;

#pragma unroll
  for (int kc = 0; kc < 8; ++kc) {   // K = 2F = 256, chunks of 32
    const _Float16* src = (kc < 4) ? (xrow + kc * 32) : (crow + (kc - 4) * 32);
    v8h lo = *(const v8h*)(src + 16 * h);
    v8h hi = *(const v8h*)(src + 16 * h + 8);
    v16h a = join8(lo, hi);
#pragma unroll
    for (int et = 0; et < 8; ++et) {
      v16h bw = *(const v16h*)(Wh + (size_t)(et * 16 + l16) * (2 * F_) + kc * 32 + 16 * h);
      acc[et] = wmma_f16(a, bw, acc[et]);
    }
  }

  float* ob = out + (size_t)b * N_ * E_;
#pragma unroll
  for (int et = 0; et < 8; ++et) {
#pragma unroll
    for (int v = 0; v < 8; ++v) {
      float r = acc[et][v];
      ob[(size_t)(row0 + v + 8 * h) * E_ + et * 16 + l16] = r > 0.0f ? r : 0.0f;
    }
  }
}

extern "C" void kernel_launch(void* const* d_in, const int* in_sizes, int n_in,
                              void* d_out, int out_size, void* d_ws, size_t ws_size,
                              hipStream_t stream) {
  (void)in_sizes; (void)n_in; (void)out_size; (void)ws_size;
  const float* X   = (const float*)d_in[0];
  // d_in[1] = nodes (int64 arange) — unused by the math
  const int*   adj = (const int*)d_in[2];
  const float* W   = (const float*)d_in[3];

  const size_t nX = (size_t)B_ * N_ * F_;   // 4,194,304 elements
  _Float16* Xh = (_Float16*)d_ws;           //  8 MB
  _Float16* Xt = Xh + nX;                   //  8 MB
  _Float16* Ch = Xt + nX;                   //  8 MB
  _Float16* Wh = Ch + nX;                   // 64 KB
  float* out = (float*)d_out;

  cvt_x_kernel<<<(unsigned)((nX + 255) / 256), 256, 0, stream>>>(X, Xh, Xt);
  cvt_w_kernel<<<(E_ * 2 * F_ + 255) / 256, 256, 0, stream>>>(W, Wh);

  const int nWaves = B_ * N_ / 16;          // 2048 row blocks
  attn_kernel<<<nWaves / 8, 256, 0, stream>>>(Xh, Xt, adj, Ch);
  linear_kernel<<<nWaves / 8, 256, 0, stream>>>(Xh, Ch, Wh, out);
}
